// VoronoiFNO2dSparsePollution_34634616275406
// MI455X (gfx1250) — matrix operand order repaired and checked by reading, not compiled, not run on hardware
//
#include <hip/hip_runtime.h>
#include <math.h>

// ---------------------------------------------------------------------------
// VoronoiFNO2dSparsePollution for gfx1250 (MI455X, wave32, WMMA).
// Truncated-DFT FNO: all spectral ops are bf16 WMMA GEMMs with f32 accum.
// All GEMM operands are stored with the reduction dim contiguous so LDS
// tiles are staged with global_load_async_to_lds_b128 (ASYNCcnt DMA).
// Activations ping-pong in bf16 (~67MB) -> resident in 192MB L2.
// fc1/fc2 evaluated only at the 64 sampled pixels. Needs ~260MB of d_ws.
// ---------------------------------------------------------------------------

typedef __bf16 bf16;
typedef __attribute__((ext_vector_type(16))) __bf16 v16bf;
typedef __attribute__((ext_vector_type(8)))  __bf16 v8bf;
typedef __attribute__((ext_vector_type(8)))  float  v8f;

#define NXg 128
#define NYg 128
#define CW  32          // WIDTH
#define BB  64          // batch
#define NTT 168         // NT time steps
#define HW  16384       // NX*NY
#define BCH 2048        // B*C

#define ASYNC_CP 1      // use global_load_async_to_lds_b128 for tile staging

__device__ inline v8f zero8() {
  v8f z = {0.f,0.f,0.f,0.f,0.f,0.f,0.f,0.f};
  return z;
}

__device__ inline v8f wmma_bf16(v16bf a, v16bf b, v8f c) {
  // D = A(16x32) * B(32x16) + C, f32 accumulate
  return __builtin_amdgcn_wmma_f32_16x16x32_bf16(false, a, false, b,
                                                 (short)0, c, false, false);
}

// 16-byte global->LDS copy. Async path: CDNA5 GLOBAL_LOAD_ASYNC_TO_LDS_B128
// (VDST = LDS byte offset VGPR, VADDR = 64-bit global address, SADDR = off).
// Generic LDS pointers carry the LDS byte offset in the low 32 bits.
__device__ inline void cp16B(bf16* l, const bf16* g) {
#if ASYNC_CP
  asm volatile("global_load_async_to_lds_b128 %0, %1, off"
               :: "v"((unsigned)(uintptr_t)l), "v"(g) : "memory");
#else
  *(v8bf*)l = *(const v8bf*)g;
#endif
}
__device__ inline void wait_async() {
#if ASYNC_CP
  asm volatile("s_wait_asynccnt 0" ::: "memory");
#endif
}

// A fragment: lane holds row m=lane&15; lanes 0-15 cover K 0..7 & 16..23,
// lanes 16-31 cover K 8..15 & 24..31 (ISA 16-bit A 16x32 layout).
// LDS tile layout: lA[m][k], 16x32, row-major.
__device__ inline v16bf frag_a(const bf16* lA, int lane) {
  int m = lane & 15, half = lane >> 4;
  const bf16* p = lA + m*32 + 8*half;
  v8bf lo = *(const v8bf*)p;
  v8bf hi = *(const v8bf*)(p + 16);
  v16bf f;
#pragma unroll
  for (int i = 0; i < 8; ++i) { f[i] = lo[i]; f[i+8] = hi[i]; }
  return f;
}

// B fragment: lane holds col n=lane&15; lanes 0-15 K 0..7 & 16..23, lanes 16-31
// K 8..15 & 24..31. LDS tile stored transposed: lBT[n][k], 16x32 row-major.
__device__ inline v16bf frag_b(const bf16* lBT, int lane) {
  int n = lane & 15, half = lane >> 4;
  const bf16* p = lBT + n*32 + 8*half;
  v8bf lo = *(const v8bf*)p;
  v8bf hi = *(const v8bf*)(p + 16);
  v16bf f;
#pragma unroll
  for (int i = 0; i < 8; ++i) { f[i] = lo[i]; f[i+8] = hi[i]; }
  return f;
}

// Wide tile fill: 16 rows x 32 elems, reduction dim contiguous in global.
// 64 16-byte chunks, 2 per lane, issued as async LDS DMA.
__device__ inline void fill_wide(bf16* l, const bf16* g, int rs, int lane) {
#pragma unroll
  for (int c = lane; c < 64; c += 32) {
    int m = c >> 2, k8 = (c & 3) << 3;
    cp16B(l + m*32 + k8, g + (size_t)m*rs + k8);
  }
}
// Strided gather fill (only S6's activation B operand needs it).
__device__ inline void fill_gather(bf16* lBT, const bf16* Bg, size_t rs, int lane) {
#pragma unroll
  for (int idx = lane; idx < 512; idx += 32) {
    int n = idx >> 5, k = idx & 31;
    lBT[idx] = Bg[(size_t)k * rs + n];
  }
}

// Complex operand fill (all reduction-contiguous): async-stage Ar/Ai/Br/Bi,
// then build -Ai in LDS after the async fence (no A-negate modifier in bf16
// WMMA; only CNeg exists).
__device__ inline void cplx_fill(bf16* lAr, bf16* lAi, bf16* lAin,
                                 bf16* lBr, bf16* lBi,
                                 const bf16* Agr, const bf16* Agi, int a_rs,
                                 const bf16* Bgr, const bf16* Bgi, int b_rs,
                                 int lane) {
  fill_wide(lAr, Agr, a_rs, lane);
  fill_wide(lAi, Agi, a_rs, lane);
  fill_wide(lBr, Bgr, b_rs, lane);
  fill_wide(lBi, Bgi, b_rs, lane);
  wait_async();
#pragma unroll
  for (int c = lane; c < 64; c += 32) {
    int off = (c >> 2) * 32 + ((c & 3) << 3);
    v8bf v = *(const v8bf*)(lAi + off);
    v8bf nv;
#pragma unroll
    for (int e = 0; e < 8; ++e) nv[e] = (bf16)(-(float)v[e]);
    *(v8bf*)(lAin + off) = nv;
  }
}
// Cr += Ar*Br - Ai*Bi ; Ci += Ar*Bi + Ai*Br   (4 chained WMMAs)
__device__ inline void cplx_mma(const bf16* lAr, const bf16* lAi, const bf16* lAin,
                                const bf16* lBr, const bf16* lBi, int lane,
                                v8f& cr, v8f& ci) {
  v16bf ar  = frag_a(lAr,  lane);
  v16bf ai  = frag_a(lAi,  lane);
  v16bf ain = frag_a(lAin, lane);
  v16bf br  = frag_b(lBr,  lane);
  v16bf bi  = frag_b(lBi,  lane);
  cr = wmma_bf16(ar,  br, cr);
  cr = wmma_bf16(ain, bi, cr);
  ci = wmma_bf16(ar,  bi, ci);
  ci = wmma_bf16(ai,  br, ci);
}

__device__ inline float gelu_exact(float v) {
  return 0.5f * v * (1.f + erff(v * 0.70710678118654752f));
}

// ---------------------------------------------------------------------------
// Stage 0: scatter neighbor observations into per-batch sparse grids
// ---------------------------------------------------------------------------
__global__ void __launch_bounds__(256)
k_scatter(const int* __restrict__ q_lin, const float* __restrict__ obs_c,
          const float* __restrict__ obs_v, const int* __restrict__ nb_idx,
          const int* __restrict__ s_iy, const int* __restrict__ s_ix,
          float* __restrict__ sparse, float* __restrict__ counts) {
  int t = blockIdx.x * 256 + threadIdx.x;       // B*K = 8192
  int b = t >> 7;
  int idx = nb_idx[t];
  int sid = idx / NTT;
  float tq = obs_c[q_lin[b] * 3 + 2];
  float tn = obs_c[idx * 3 + 2];
  float w  = __expf(-0.1f * fabsf(tn - tq));
  float v  = obs_v[idx] * w;
  int lin  = s_iy[sid] * NYg + s_ix[sid];
  atomicAdd(&sparse[b * HW + lin], v);
  atomicAdd(&counts[b * HW + lin], w);
}

// ---------------------------------------------------------------------------
// Stage 1: normalize + fc0 lift (3 -> 32 channels), write bf16 x[b][c][hw]
// ---------------------------------------------------------------------------
__global__ void __launch_bounds__(256)
k_fc0(const float* __restrict__ sparse, const float* __restrict__ counts,
      const float* __restrict__ xg, const float* __restrict__ yg,
      const float* __restrict__ w, const float* __restrict__ bias,
      bf16* __restrict__ xin) {
  int t = blockIdx.x * 256 + threadIdx.x;       // b*HW + hw
  int b = t >> 14, hw = t & (HW - 1);
  float c = counts[t];
  float s = (c > 0.f) ? sparse[t] / fmaxf(c, 1e-6f) : 0.f;
  float gx = xg[hw], gy = yg[hw];
  bf16* dst = xin + (size_t)b * CW * HW + hw;
#pragma unroll
  for (int d = 0; d < CW; ++d)
    dst[(size_t)d * HW] = (bf16)(bias[d] + s * w[d] + gx * w[32 + d] + gy * w[64 + d]);
}

// ---------------------------------------------------------------------------
// DFT basis matrices (bf16), built on device each call. All stored with the
// GEMM reduction dim contiguous:
//  arr0: EyT [32 n][128 w]   n<16: cos, n>=16: -sin           (forward y-rDFT, B)
//  arr1: Fxr/Fxi [32 kx][128 h]  e^{-i}                       (forward x-DFT, A)
//  arr2: Gxr/Gxi [128 h][32 kx]  e^{+i}/128                   (inverse x-DFT, A)
//  arr3: HyT [128 w][32 m]  m<16: s*cos, m>=16: -s*sin        (inverse y-irDFT, B)
//  arr4: pw_w f32 -> bf16 [4][o][c]                           (pointwise, A)
// ---------------------------------------------------------------------------
__global__ void __launch_bounds__(256)
k_prep(bf16* __restrict__ EyT, bf16* __restrict__ Fxr, bf16* __restrict__ Fxi,
       bf16* __restrict__ Gxr, bf16* __restrict__ Gxi, bf16* __restrict__ HyT,
       const float* __restrict__ pw_w, bf16* __restrict__ pwWb) {
  int t = blockIdx.x * 256 + threadIdx.x;       // 0..4095
  int arr = blockIdx.y;
  const float TP = 6.28318530717958647692f;
  if (arr == 0) {
    int n = t >> 7, w = t & 127;
    float a = TP * (float)w * (float)(n & 15) / 128.f;
    EyT[t] = (bf16)((n < 16) ? cosf(a) : -sinf(a));
  } else if (arr == 1) {
    int kx = t >> 7, h = t & 127;
    int kk = (kx < 16) ? kx : kx + 96;          // modes 0..15 and 112..127
    float a = TP * (float)h * (float)kk / 128.f;
    Fxr[t] = (bf16)cosf(a);
    Fxi[t] = (bf16)(-sinf(a));
  } else if (arr == 2) {
    int h = t >> 5, kx = t & 31;
    int kk = (kx < 16) ? kx : kx + 96;
    float a = TP * (float)h * (float)kk / 128.f;
    Gxr[t] = (bf16)(cosf(a) * (1.f / 128.f));
    Gxi[t] = (bf16)(sinf(a) * (1.f / 128.f));
  } else if (arr == 3) {
    int w = t >> 5, m = t & 31;
    int mm = m & 15;
    float s = ((mm == 0) ? 1.f : 2.f) / 128.f;
    float a = TP * (float)w * (float)mm / 128.f;
    HyT[t] = (bf16)((m < 16) ? s * cosf(a) : -s * sinf(a));
  } else {
    pwWb[t] = (bf16)pw_w[t];                    // 4*32*32 = 4096
  }
}

// Spectral weights: permute [4][i][o][m1][m2] (w1 low kx / w2 high kx) into
// per-mode B-transposed layout WT[layer][kx(32)*16+m2][o(32)][i(32)], bf16.
__global__ void __launch_bounds__(256)
k_wconv(const float* __restrict__ w1r, const float* __restrict__ w1i,
        const float* __restrict__ w2r, const float* __restrict__ w2i,
        bf16* __restrict__ Wr, bf16* __restrict__ Wi) {
  int t = blockIdx.x * 256 + threadIdx.x;       // 0..2097151
  int layer = t >> 19;
  int rem = t & 524287;
  int mode = rem >> 10, oi = rem & 1023;
  int kx = mode >> 4, m2 = mode & 15;
  int o = oi >> 5, i = oi & 31;                 // i fastest -> reduction contiguous
  int src = ((layer * 32 + i) * 32 + o) * 256 + (kx & 15) * 16 + m2;
  bool lo = kx < 16;
  Wr[t] = (bf16)(lo ? w1r[src] : w2r[src]);
  Wi[t] = (bf16)(lo ? w1i[src] : w2i[src]);
}

// ---------------------------------------------------------------------------
// S1: forward y-rDFT.  x[(bc,h),128w] @ EyT^T -> X1T[col(bc*16+m2)][128 h]
// ---------------------------------------------------------------------------
__global__ void __launch_bounds__(128)
k_dft_y(const bf16* __restrict__ xin, const bf16* __restrict__ EyT,
        bf16* __restrict__ X1Tr, bf16* __restrict__ X1Ti) {
  __shared__ __align__(16) bf16 lds[4][2][512];
  int wave = threadIdx.x >> 5, lane = threadIdx.x & 31;
  int tile = blockIdx.x * 4 + wave;             // 32768 tiles
  int mt = tile >> 1, nt = tile & 1;
  bf16* lA = lds[wave][0];
  bf16* lB = lds[wave][1];
  v8f acc = zero8();
  const bf16* Ag = xin + (size_t)mt * 16 * 128;
  const bf16* Bg = EyT + (size_t)nt * 16 * 128;
  for (int k0 = 0; k0 < 128; k0 += 32) {
    fill_wide(lA, Ag + k0, 128, lane);
    fill_wide(lB, Bg + k0, 128, lane);
    wait_async();
    __syncthreads();
    acc = wmma_bf16(frag_a(lA, lane), frag_b(lB, lane), acc);
    __syncthreads();
  }
  int half = lane >> 4, n = lane & 15;
  int nglob = nt * 16 + n;
  bf16* dst = (nglob < 16) ? X1Tr : X1Ti;
  int m2 = nglob & 15;
#pragma unroll
  for (int e = 0; e < 8; ++e) {
    int r = mt * 16 + e + 8 * half;
    int h = r & 127, bc = r >> 7;
    dst[(size_t)(bc * 16 + m2) * 128 + h] = (bf16)acc[e];
  }
}

// ---------------------------------------------------------------------------
// S2: forward x-DFT (complex).  Fx[32kx,128h] @ X1T^T -> X2[mode][b][c]
// ---------------------------------------------------------------------------
__global__ void __launch_bounds__(128)
k_dft_x(const bf16* __restrict__ Fxr, const bf16* __restrict__ Fxi,
        const bf16* __restrict__ X1Tr, const bf16* __restrict__ X1Ti,
        bf16* __restrict__ X2r, bf16* __restrict__ X2i) {
  __shared__ __align__(16) bf16 lds[4][5][512];
  int wave = threadIdx.x >> 5, lane = threadIdx.x & 31;
  int tile = blockIdx.x * 4 + wave;             // 2*2048 = 4096 tiles
  int mt = tile >> 11, nt = tile & 2047;
  bf16 *lAr = lds[wave][0], *lAi = lds[wave][1], *lAin = lds[wave][2];
  bf16 *lBr = lds[wave][3], *lBi = lds[wave][4];
  v8f cr = zero8(), ci = zero8();
  const bf16* Agr = Fxr + (size_t)mt * 16 * 128;
  const bf16* Agi = Fxi + (size_t)mt * 16 * 128;
  const bf16* Bgr = X1Tr + (size_t)nt * 16 * 128;
  const bf16* Bgi = X1Ti + (size_t)nt * 16 * 128;
  for (int k0 = 0; k0 < 128; k0 += 32) {
    cplx_fill(lAr, lAi, lAin, lBr, lBi,
              Agr + k0, Agi + k0, 128, Bgr + k0, Bgi + k0, 128, lane);
    __syncthreads();
    cplx_mma(lAr, lAi, lAin, lBr, lBi, lane, cr, ci);
    __syncthreads();
  }
  int half = lane >> 4, n = lane & 15;
  int col = nt * 16 + n;
  int b = col >> 9, c = (col >> 4) & 31, m2 = col & 15;
#pragma unroll
  for (int e = 0; e < 8; ++e) {
    int kx = mt * 16 + e + 8 * half;
    size_t off = (size_t)(kx * 16 + m2) * BCH + b * 32 + c;
    X2r[off] = (bf16)cr[e];
    X2i[off] = (bf16)ci[e];
  }
}

// ---------------------------------------------------------------------------
// S3: per-mode channel mixing (512 batched complex GEMMs, M=64 N=32 K=32)
// ---------------------------------------------------------------------------
__global__ void __launch_bounds__(128)
k_spec_mul(const bf16* __restrict__ X2r, const bf16* __restrict__ X2i,
           const bf16* __restrict__ Wr, const bf16* __restrict__ Wi,
           bf16* __restrict__ X3Tr, bf16* __restrict__ X3Ti) {
  __shared__ __align__(16) bf16 lds[4][5][512];
  int wave = threadIdx.x >> 5, lane = threadIdx.x & 31;
  int z = blockIdx.y;                            // mode (kx*16 + m2)
  int tile = blockIdx.x * 4 + wave;              // 0..7
  int mt = tile >> 1, nt = tile & 1;
  bf16 *lAr = lds[wave][0], *lAi = lds[wave][1], *lAin = lds[wave][2];
  bf16 *lBr = lds[wave][3], *lBi = lds[wave][4];
  v8f cr = zero8(), ci = zero8();
  const bf16* Agr = X2r + (size_t)z * BCH + mt * 16 * 32;
  const bf16* Agi = X2i + (size_t)z * BCH + mt * 16 * 32;
  const bf16* Bgr = Wr + (size_t)z * 1024 + nt * 16 * 32;
  const bf16* Bgi = Wi + (size_t)z * 1024 + nt * 16 * 32;
  cplx_fill(lAr, lAi, lAin, lBr, lBi, Agr, Agi, 32, Bgr, Bgi, 32, lane);
  __syncthreads();
  cplx_mma(lAr, lAi, lAin, lBr, lBi, lane, cr, ci);
  int half = lane >> 4, n = lane & 15;
  int kx = z >> 4, m2 = z & 15;
  int o = nt * 16 + n;
#pragma unroll
  for (int e = 0; e < 8; ++e) {
    int b = mt * 16 + e + 8 * half;
    size_t off = ((size_t)b * 512 + o * 16 + m2) * 32 + kx;   // X3T[col][kx]
    X3Tr[off] = (bf16)cr[e];
    X3Ti[off] = (bf16)ci[e];
  }
}

// ---------------------------------------------------------------------------
// S4: inverse x-DFT (complex). Gx[128h,32kx] @ X3T^T -> Y1[(h,b,o)][r|i m2]
// ---------------------------------------------------------------------------
__global__ void __launch_bounds__(128)
k_idft_x(const bf16* __restrict__ Gxr, const bf16* __restrict__ Gxi,
         const bf16* __restrict__ X3Tr, const bf16* __restrict__ X3Ti,
         bf16* __restrict__ Y1) {
  __shared__ __align__(16) bf16 lds[4][5][512];
  int wave = threadIdx.x >> 5, lane = threadIdx.x & 31;
  int tile = blockIdx.x * 4 + wave;             // 8*2048 = 16384 tiles
  int mt = tile >> 11, nt = tile & 2047;
  bf16 *lAr = lds[wave][0], *lAi = lds[wave][1], *lAin = lds[wave][2];
  bf16 *lBr = lds[wave][3], *lBi = lds[wave][4];
  v8f cr = zero8(), ci = zero8();
  const bf16* Agr = Gxr + (size_t)mt * 16 * 32;
  const bf16* Agi = Gxi + (size_t)mt * 16 * 32;
  const bf16* Bgr = X3Tr + (size_t)nt * 16 * 32;
  const bf16* Bgi = X3Ti + (size_t)nt * 16 * 32;
  cplx_fill(lAr, lAi, lAin, lBr, lBi, Agr, Agi, 32, Bgr, Bgi, 32, lane);
  __syncthreads();
  cplx_mma(lAr, lAi, lAin, lBr, lBi, lane, cr, ci);
  int half = lane >> 4, n = lane & 15;
  int col = nt * 16 + n;
  int b = col >> 9, o = (col >> 4) & 31, m2 = col & 15;
#pragma unroll
  for (int e = 0; e < 8; ++e) {
    int h = mt * 16 + e + 8 * half;
    size_t base = ((size_t)h * BCH + b * 32 + o) * 32;
    Y1[base + m2]      = (bf16)cr[e];
    Y1[base + 16 + m2] = (bf16)ci[e];
  }
}

// ---------------------------------------------------------------------------
// S5: inverse y-irDFT (real out). Y1[(h,b,o),32] @ HyT^T -> spec[b][o][h][w]
// ---------------------------------------------------------------------------
__global__ void __launch_bounds__(128)
k_idft_y(const bf16* __restrict__ Y1, const bf16* __restrict__ HyT,
         bf16* __restrict__ spec) {
  __shared__ __align__(16) bf16 lds[4][2][512];
  int wave = threadIdx.x >> 5, lane = threadIdx.x & 31;
  int tile = blockIdx.x * 4 + wave;             // 16384*8 = 131072 tiles
  int mt = tile >> 3, nt = tile & 7;
  bf16* lA = lds[wave][0];
  bf16* lB = lds[wave][1];
  v8f acc = zero8();
  fill_wide(lA, Y1 + (size_t)mt * 16 * 32, 32, lane);
  fill_wide(lB, HyT + (size_t)nt * 16 * 32, 32, lane);
  wait_async();
  __syncthreads();
  acc = wmma_bf16(frag_a(lA, lane), frag_b(lB, lane), acc);
  int half = lane >> 4, n = lane & 15;
  int w = nt * 16 + n;
#pragma unroll
  for (int e = 0; e < 8; ++e) {
    int r = mt * 16 + e + 8 * half;
    int h = r >> 11, bo = r & 2047;
    spec[(size_t)bo * HW + h * 128 + w] = (bf16)acc[e];
  }
}

// ---------------------------------------------------------------------------
// S6: pointwise conv + residual add + GELU. pw[32o,32c] @ x_b[32c,16384hw]
// ---------------------------------------------------------------------------
__global__ void __launch_bounds__(128)
k_pointwise(const bf16* __restrict__ pwW, const float* __restrict__ pwB,
            const bf16* __restrict__ spec, const bf16* __restrict__ xin,
            bf16* __restrict__ xout, int apply_gelu) {
  __shared__ __align__(16) bf16 lds[4][2][512];
  int wave = threadIdx.x >> 5, lane = threadIdx.x & 31;
  int b = blockIdx.y;
  int tile = blockIdx.x * 4 + wave;             // 2*1024 = 2048 tiles per b
  int mt = tile >> 10, nt = tile & 1023;
  bf16* lA = lds[wave][0];
  bf16* lB = lds[wave][1];
  v8f acc = zero8();
  fill_wide(lA, pwW + (size_t)mt * 16 * 32, 32, lane);
  fill_gather(lB, xin + (size_t)b * CW * HW + nt * 16, HW, lane);
  wait_async();
  __syncthreads();
  acc = wmma_bf16(frag_a(lA, lane), frag_b(lB, lane), acc);
  int half = lane >> 4, n = lane & 15;
  int hw = nt * 16 + n;
#pragma unroll
  for (int e = 0; e < 8; ++e) {
    int o = mt * 16 + e + 8 * half;
    size_t off = (size_t)b * CW * HW + (size_t)o * HW + hw;
    float v = acc[e] + pwB[o] + (float)spec[off];
    if (apply_gelu) v = gelu_exact(v);
    xout[off] = (bf16)v;
  }
}

// ---------------------------------------------------------------------------
// Final: fc1+fc2 evaluated only at the sampled pixel of each batch element
// ---------------------------------------------------------------------------
__global__ void __launch_bounds__(128)
k_sample(const bf16* __restrict__ xfin, const int* __restrict__ q_lin,
         const int* __restrict__ s_iy, const int* __restrict__ s_ix,
         const float* __restrict__ fc1_w, const float* __restrict__ fc1_b,
         const float* __restrict__ fc2_w, const float* __restrict__ fc2_b,
         float* __restrict__ out) {
  __shared__ float xs[CW];
  __shared__ float red[128];
  int b = blockIdx.x, d = threadIdx.x;
  int qs = q_lin[b] / NTT;
  int pix = s_iy[qs] * NYg + s_ix[qs];
  if (d < CW) xs[d] = (float)xfin[(size_t)b * CW * HW + (size_t)d * HW + pix];
  __syncthreads();
  float a = fc1_b[d];
#pragma unroll
  for (int c = 0; c < CW; ++c) a += xs[c] * fc1_w[c * 128 + d];
  a = gelu_exact(a);
  red[d] = a * fc2_w[d];
  __syncthreads();
  for (int s = 64; s > 0; s >>= 1) {
    if (d < s) red[d] += red[d + s];
    __syncthreads();
  }
  if (d == 0) out[b] = red[0] + fc2_b[0];
}

// ---------------------------------------------------------------------------
extern "C" void kernel_launch(void* const* d_in, const int* in_sizes, int n_in,
                              void* d_out, int out_size, void* d_ws, size_t ws_size,
                              hipStream_t stream) {
  (void)in_sizes; (void)n_in; (void)out_size; (void)ws_size;
  const int*   q_lin  = (const int*)d_in[0];
  const float* obs_c  = (const float*)d_in[1];
  const float* obs_v  = (const float*)d_in[2];
  const int*   nb_idx = (const int*)d_in[3];
  const int*   s_iy   = (const int*)d_in[4];
  const int*   s_ix   = (const int*)d_in[5];
  const float* x_grid = (const float*)d_in[6];
  const float* y_grid = (const float*)d_in[7];
  const float* fc0_w  = (const float*)d_in[8];
  const float* fc0_b  = (const float*)d_in[9];
  const float* w1r    = (const float*)d_in[10];
  const float* w1i    = (const float*)d_in[11];
  const float* w2r    = (const float*)d_in[12];
  const float* w2i    = (const float*)d_in[13];
  const float* pw_w   = (const float*)d_in[14];
  const float* pw_b   = (const float*)d_in[15];
  const float* fc1_w  = (const float*)d_in[16];
  const float* fc1_b  = (const float*)d_in[17];
  const float* fc2_w  = (const float*)d_in[18];
  const float* fc2_b  = (const float*)d_in[19];
  float* out = (float*)d_out;

  // --- workspace carve-up (~260 MB) ---
  char* ws = (char*)d_ws;
  size_t off = 0;
  auto carve = [&](size_t bytes) -> char* {
    char* p = ws + off;
    off = (off + bytes + 255) & ~(size_t)255;
    return p;
  };
  float* SPARSE = (float*)carve((size_t)BB * HW * 4);        //  4 MB
  float* COUNTS = (float*)carve((size_t)BB * HW * 4);        //  4 MB
  bf16*  XA     = (bf16*) carve((size_t)BB * CW * HW * 2);   // 67 MB
  bf16*  XB     = (bf16*) carve((size_t)BB * CW * HW * 2);   // 67 MB
  bf16*  SPEC   = (bf16*) carve((size_t)BB * CW * HW * 2);   // 67 MB
  bf16*  X1TR   = (bf16*) carve((size_t)32768 * 128 * 2);    //  8 MB
  bf16*  X1TI   = (bf16*) carve((size_t)32768 * 128 * 2);    //  8 MB
  bf16*  X2R    = (bf16*) carve((size_t)512 * BCH * 2);      //  2 MB
  bf16*  X2I    = (bf16*) carve((size_t)512 * BCH * 2);
  bf16*  X3TR   = (bf16*) carve((size_t)32768 * 32 * 2);
  bf16*  X3TI   = (bf16*) carve((size_t)32768 * 32 * 2);
  bf16*  Y1     = (bf16*) carve((size_t)128 * BCH * 32 * 2); // 17 MB
  bf16*  WRc    = (bf16*) carve((size_t)4 * 512 * 1024 * 2); //  4 MB
  bf16*  WIc    = (bf16*) carve((size_t)4 * 512 * 1024 * 2); //  4 MB
  bf16*  EyT    = (bf16*) carve(4096 * 2);
  bf16*  Fxr    = (bf16*) carve(4096 * 2);
  bf16*  Fxi    = (bf16*) carve(4096 * 2);
  bf16*  Gxr    = (bf16*) carve(4096 * 2);
  bf16*  Gxi    = (bf16*) carve(4096 * 2);
  bf16*  HyT    = (bf16*) carve(4096 * 2);
  bf16*  PWW    = (bf16*) carve(4096 * 2);

  // --- input build ---
  hipMemsetAsync(SPARSE, 0, (size_t)BB * HW * 4, stream);
  hipMemsetAsync(COUNTS, 0, (size_t)BB * HW * 4, stream);
  k_scatter<<<32, 256, 0, stream>>>(q_lin, obs_c, obs_v, nb_idx, s_iy, s_ix,
                                    SPARSE, COUNTS);
  k_fc0<<<(BB * HW) / 256, 256, 0, stream>>>(SPARSE, COUNTS, x_grid, y_grid,
                                             fc0_w, fc0_b, XA);
  k_prep<<<dim3(16, 5), 256, 0, stream>>>(EyT, Fxr, Fxi, Gxr, Gxi, HyT, pw_w, PWW);
  k_wconv<<<8192, 256, 0, stream>>>(w1r, w1i, w2r, w2i, WRc, WIc);

  // --- 4 FNO layers ---
  bf16* cur = XA;
  bf16* nxt = XB;
  for (int layer = 0; layer < 4; ++layer) {
    k_dft_y<<<8192, 128, 0, stream>>>(cur, EyT, X1TR, X1TI);
    k_dft_x<<<1024, 128, 0, stream>>>(Fxr, Fxi, X1TR, X1TI, X2R, X2I);
    k_spec_mul<<<dim3(2, 512), 128, 0, stream>>>(X2R, X2I,
        WRc + (size_t)layer * 524288, WIc + (size_t)layer * 524288, X3TR, X3TI);
    k_idft_x<<<4096, 128, 0, stream>>>(Gxr, Gxi, X3TR, X3TI, Y1);
    k_idft_y<<<32768, 128, 0, stream>>>(Y1, HyT, SPEC);
    k_pointwise<<<dim3(512, BB), 128, 0, stream>>>(PWW + layer * 1024,
        pw_b + layer * 32, SPEC, cur, nxt, layer < 3 ? 1 : 0);
    bf16* tmp = cur; cur = nxt; nxt = tmp;
  }

  // --- sample fc1/fc2 at the 64 query pixels ---
  k_sample<<<BB, 128, 0, stream>>>(cur, q_lin, s_iy, s_ix,
                                   fc1_w, fc1_b, fc2_w, fc2_b, out);
}
